// DCEMAE_78889959293298
// MI455X (gfx1250) — compile-verified, matrix operand order checked by reading, not run on hardware
//
#include <hip/hip_runtime.h>

// ---------------------------------------------------------------------------
// MI455X (gfx1250) implementation of the dual-branch transformer classifier.
// fp32 tensors in global memory; bf16 conversion during LDS tile fill;
// fp32 accumulation via v_wmma_f32_16x16x32_bf16.  wave32 / WGP layout.
// ---------------------------------------------------------------------------

typedef __attribute__((ext_vector_type(16))) __bf16 v16bf;
typedef __attribute__((ext_vector_type(2)))  __bf16 v2bf;
typedef __attribute__((ext_vector_type(8)))  float  v8f;

struct __attribute__((aligned(16))) U128 { unsigned int w[4]; };
union Frag { v16bf v; U128 q[2]; };

__device__ __forceinline__ unsigned int pack2bf(float lo, float hi) {
#if __has_builtin(__builtin_amdgcn_cvt_pk_bf16_f32)
    v2bf p = __builtin_amdgcn_cvt_pk_bf16_f32(lo, hi);
    return __builtin_bit_cast(unsigned int, p);
#else
    v2bf p;
    p.x = (__bf16)lo;                    // clang lowers to v_cvt bf16 ops
    p.y = (__bf16)hi;
    return __builtin_bit_cast(unsigned int, p);
#endif
}

// ---------------------------------------------------------------------------
// Generic batched GEMM:  C = op( alpha * A x B(^T) + bias )
//   A: [M,K] (lda), B: [K,N] (bT=0) or [N,K] (bT=1);  z = zo*Hinner + zi
//   REQUIRES: M % 128 == 0, K % 64 == 0, row bases 16-byte aligned
//             (all true for every GEMM in this network; N may be ragged).
// Block: 256 threads = 8 waves (4 x 2), tile BM=128 BN=128 BK=64.
// Each wave: 32x64 patch = 2x4 wmma tiles, 2 k-steps per LDS tile.
// ---------------------------------------------------------------------------
__global__ __launch_bounds__(256)
void gemm_bf16_wmma(const float* __restrict__ A, int lda, long long sAo, long long sAi,
                    const float* __restrict__ Bm, int ldb, long long sBo, long long sBi, int bT,
                    float* __restrict__ C, int ldc, long long sCo, long long sCi,
                    const float* __restrict__ bias, float alpha, int gelu,
                    int N, int K, int Hinner)
{
    const int z  = blockIdx.z;
    const int zo = z / Hinner, zi = z % Hinner;
    A  += zo * sAo + zi * sAi;
    Bm += zo * sBo + zi * sBi;
    C  += zo * sCo + zi * sCi;

    __shared__ __attribute__((aligned(16))) unsigned short As[128][72]; // bf16 bits, +pad
    __shared__ __attribute__((aligned(16))) unsigned short Bs[128][72]; // stored [n][k]

    const int tid   = threadIdx.x;
    const int lane  = tid & 31;
    const int wave  = tid >> 5;
    const int waveM = wave & 3;          // 0..3 -> 32-row bands
    const int waveN = wave >> 2;         // 0..1 -> 64-col bands
    const int rowbase = blockIdx.y * 128 + waveM * 32;
    const int colbase = blockIdx.x * 128 + waveN * 64;

    v8f acc[2][4] = {};

    // tile fill: 2 threads per row, 32 consecutive k each
    const int ldrow = tid >> 1;
    const int lks   = (tid & 1) << 5;              // 0 or 32
    const int gy = blockIdx.y * 128 + ldrow;       // A row (always < M)
    const int gx = blockIdx.x * 128 + ldrow;       // B row (bT) / col (!bT)

    // fragment addressing (ISA 16-bit A/B layouts, wave32)
    const int lrow = lane & 15;
    const int kb   = (lane < 16) ? 0 : 8;

    for (int k0 = 0; k0 < K; k0 += 64) {
        // ---- gather 32 A values (vectorized b128 loads, no guard: M%128==0) --
        float av[32];
        {
            const float4* ap = (const float4*)(A + (long long)gy * lda + k0 + lks);
            #pragma unroll
            for (int c = 0; c < 8; ++c) {
                const float4 t = ap[c];
                av[4*c+0] = t.x; av[4*c+1] = t.y; av[4*c+2] = t.z; av[4*c+3] = t.w;
            }
        }
        // ---- gather 32 B values (N may be ragged -> one row guard) ----------
        float bv[32];
        if (gx < N) {
            if (bT) {
                const float4* bp = (const float4*)(Bm + (long long)gx * ldb + k0 + lks);
                #pragma unroll
                for (int c = 0; c < 8; ++c) {
                    const float4 t = bp[c];
                    bv[4*c+0] = t.x; bv[4*c+1] = t.y; bv[4*c+2] = t.z; bv[4*c+3] = t.w;
                }
            } else {
                #pragma unroll
                for (int e = 0; e < 32; ++e)
                    bv[e] = Bm[(long long)(k0 + lks + e) * ldb + gx];
            }
        } else {
            #pragma unroll
            for (int e = 0; e < 32; ++e) bv[e] = 0.f;
        }
        if (k0 + 64 < K)
            __builtin_prefetch(A + (long long)gy * lda + k0 + 64 + lks, 0, 1);

        // ---- pack to bf16 pairs, store as 16B chunks (ds_store_b128) --------
        #pragma unroll
        for (int c = 0; c < 4; ++c) {
            U128 sa, sb;
            #pragma unroll
            for (int j = 0; j < 4; ++j) {
                sa.w[j] = pack2bf(av[8*c + 2*j], av[8*c + 2*j + 1]);
                sb.w[j] = pack2bf(bv[8*c + 2*j], bv[8*c + 2*j + 1]);
            }
            *(U128*)&As[ldrow][lks + 8*c] = sa;
            *(U128*)&Bs[ldrow][lks + 8*c] = sb;
        }
        __syncthreads();

        // ---- two WMMA k-steps per LDS tile ----------------------------------
        #pragma unroll
        for (int ks = 0; ks < 2; ++ks) {
            const int ko = ks * 32;
            Frag a[2], b[4];
            #pragma unroll
            for (int tm = 0; tm < 2; ++tm) {
                const unsigned short* p = &As[waveM * 32 + tm * 16 + lrow][ko];
                a[tm].q[0] = *(const U128*)(p + kb);
                a[tm].q[1] = *(const U128*)(p + kb + 16);
            }
            #pragma unroll
            for (int tn = 0; tn < 4; ++tn) {
                const unsigned short* p = &Bs[waveN * 64 + tn * 16 + lrow][ko];
                b[tn].q[0] = *(const U128*)(p + kb);
                b[tn].q[1] = *(const U128*)(p + kb + 16);
            }
            #pragma unroll
            for (int tm = 0; tm < 2; ++tm)
                #pragma unroll
                for (int tn = 0; tn < 4; ++tn)
                    acc[tm][tn] = __builtin_amdgcn_wmma_f32_16x16x32_bf16(
                        false, a[tm].v, false, b[tn].v, (short)0, acc[tm][tn], false, false);
        }
        __syncthreads();
    }

    // epilogue: C/D layout -> lane%16 = col, vgpr i = row (i / 8+i); M%128==0
    #pragma unroll
    for (int tm = 0; tm < 2; ++tm) {
        const int m0 = rowbase + tm * 16 + ((lane < 16) ? 0 : 8);
        #pragma unroll
        for (int tn = 0; tn < 4; ++tn) {
            const int n = colbase + tn * 16 + lrow;
            if (n < N) {
                const float bvv = bias ? bias[n] : 0.f;
                #pragma unroll
                for (int i = 0; i < 8; ++i) {
                    float val = acc[tm][tn][i] * alpha + bvv;
                    if (gelu) val = 0.5f * val * (1.0f + erff(val * 0.70710678118f));
                    C[(long long)(m0 + i) * ldc + n] = val;
                }
            }
        }
    }
}

// ---------------------------------------------------------------------------
// de-interleave x[B,L,2] -> xI[B*T,W], xQ[B*T,W]
__global__ __launch_bounds__(256)
void pack_iq_k(const float* __restrict__ x, float* __restrict__ xI,
               float* __restrict__ xQ, int n)
{
    int i = blockIdx.x * 256 + threadIdx.x;
    if (i < n) { xI[i] = x[2 * i]; xQ[i] = x[2 * i + 1]; }
}

// t[b, :, :] += pos  (t has batch stride xbs; per = T*d)
__global__ __launch_bounds__(256)
void add_pos_k(float* __restrict__ t, long long xbs,
               const float* __restrict__ pos, int per, int total)
{
    int i = blockIdx.x * 256 + threadIdx.x;
    if (i >= total) return;
    int b = i / per, r = i % per;
    t[(long long)b * xbs + r] += pos[r];
}

// wave-per-row softmax (wave32 shuffles), rows of length T (256 or 512)
__global__ __launch_bounds__(256)
void softmax_rows_k(float* __restrict__ s, int T, int nrows)
{
    int wid  = (blockIdx.x * blockDim.x + threadIdx.x) >> 5;
    int lane = threadIdx.x & 31;
    if (wid >= nrows) return;
    float* row = s + (long long)wid * T;
    const int cnt = T >> 5;                 // 8 or 16
    float vals[16];
    float mx = -3.0e38f;
    for (int i = 0; i < cnt; ++i) { vals[i] = row[lane + i * 32]; mx = fmaxf(mx, vals[i]); }
    for (int o = 16; o > 0; o >>= 1) mx = fmaxf(mx, __shfl_xor(mx, o, 32));
    float sum = 0.f;
    for (int i = 0; i < cnt; ++i) { vals[i] = expf(vals[i] - mx); sum += vals[i]; }
    for (int o = 16; o > 0; o >>= 1) sum += __shfl_xor(sum, o, 32);
    const float inv = 1.f / sum;
    for (int i = 0; i < cnt; ++i) row[lane + i * 32] = vals[i] * inv;
}

// x[row] = LayerNorm(x[row] + y[row]) * g + b   (d=512, in-place on x)
__global__ __launch_bounds__(128)
void add_ln_k(float* __restrict__ x, long long xbs, int Ttok,
              const float* __restrict__ y, const float* __restrict__ g,
              const float* __restrict__ b)
{
    constexpr int D = 512;
    const int row = blockIdx.x;
    float* xp = x + (long long)(row / Ttok) * xbs + (long long)(row % Ttok) * D;
    const float* yp = y + (long long)row * D;
    const int tid = threadIdx.x;
    float v[4], s = 0.f, s2 = 0.f;
    #pragma unroll
    for (int i = 0; i < 4; ++i) {
        const int c = tid + i * 128;
        const float u = xp[c] + yp[c];
        v[i] = u; s += u; s2 += u * u;
    }
    __shared__ float r1[128], r2[128];
    r1[tid] = s; r2[tid] = s2; __syncthreads();
    for (int o = 64; o > 0; o >>= 1) {
        if (tid < o) { r1[tid] += r1[tid + o]; r2[tid] += r2[tid + o]; }
        __syncthreads();
    }
    const float mu  = r1[0] / D;
    const float var = r2[0] / D - mu * mu;
    const float inv = rsqrtf(var + 1e-5f);
    #pragma unroll
    for (int i = 0; i < 4; ++i) {
        const int c = tid + i * 128;
        xp[c] = (v[i] - mu) * inv * g[c] + b[c];
    }
}

// m[b,c] = mean over Tt tokens of s[b, t, c]
__global__ __launch_bounds__(256)
void mean_k(const float* __restrict__ s, float* __restrict__ m,
            int Tt, long long xbs, int total)
{
    int i = blockIdx.x * 256 + threadIdx.x;
    if (i >= total) return;
    const int b = i >> 9, c = i & 511;
    const float* p = s + (long long)b * xbs + c;
    float acc = 0.f;
    for (int t = 0; t < Tt; ++t) acc += p[(long long)t * 512];
    m[i] = acc / (float)Tt;
}

// tiny linear head: [8,512]->100->50->10 (no activations)
__global__ __launch_bounds__(256)
void head_mlp_k(const float* __restrict__ m,
                const float* __restrict__ w1, const float* __restrict__ b1,
                const float* __restrict__ w2, const float* __restrict__ b2,
                const float* __restrict__ w3, const float* __restrict__ b3,
                float* __restrict__ out)
{
    __shared__ float h1[8 * 100];
    __shared__ float h2[8 * 50];
    const int tid = threadIdx.x;
    for (int i = tid; i < 800; i += 256) {
        const int b = i / 100, j = i % 100;
        float a = b1[j];
        for (int k = 0; k < 512; ++k) a += m[b * 512 + k] * w1[k * 100 + j];
        h1[i] = a;
    }
    __syncthreads();
    for (int i = tid; i < 400; i += 256) {
        const int b = i / 50, j = i % 50;
        float a = b2[j];
        for (int k = 0; k < 100; ++k) a += h1[b * 100 + k] * w2[k * 50 + j];
        h2[i] = a;
    }
    __syncthreads();
    for (int i = tid; i < 80; i += 256) {
        const int b = i / 10, j = i % 10;
        float a = b3[j];
        for (int k = 0; k < 50; ++k) a += h2[b * 50 + k] * w3[k * 10 + j];
        out[i] = a;
    }
}

// ---------------------------------------------------------------------------
// host side
// ---------------------------------------------------------------------------
static inline void G(hipStream_t st,
                     const float* A, int lda, long long sAo, long long sAi,
                     const float* B, int ldb, long long sBo, long long sBi, int bT,
                     float* C, int ldc, long long sCo, long long sCi,
                     const float* bias, float alpha, int gelu,
                     int M, int N, int K, int Hi, int Z)
{
    dim3 grid((N + 127) / 128, (M + 127) / 128, Z);
    gemm_bf16_wmma<<<grid, 256, 0, st>>>(A, lda, sAo, sAi, B, ldb, sBo, sBi, bT,
                                         C, ldc, sCo, sCi, bias, alpha, gelu,
                                         N, K, Hi);
}

extern "C" void kernel_launch(void* const* d_in, const int* in_sizes, int n_in,
                              void* d_out, int out_size, void* d_ws, size_t ws_size,
                              hipStream_t stream)
{
    (void)in_sizes; (void)n_in; (void)out_size; (void)ws_size;
    constexpr int B = 8, L = 16384, W = 64, T = 256, D = 512, DFF = 2048, NH = 8, DK = 64;
    constexpr long long SBATCH = (long long)2 * T * D;   // 262144: batch stride of s

    // ---- inputs (setup_inputs order; encoder dicts in insertion order) ----
    const float* xin = (const float*)d_in[0];
    const float* pos = (const float*)d_in[1];
    const float* cIw = (const float*)d_in[2];
    const float* cIb = (const float*)d_in[3];
    const float* cQw = (const float*)d_in[4];
    const float* cQb = (const float*)d_in[5];
    // enc leaf order: wq bq wk bk wv bv wo bo w1 bb1 w2 bb2 g1 b1 g2 b2
    const float* encI[16]; const float* encQ[16]; const float* encS[16];
    for (int i = 0; i < 16; ++i) {
        encI[i] = (const float*)d_in[6 + i];
        encQ[i] = (const float*)d_in[22 + i];
        encS[i] = (const float*)d_in[38 + i];
    }
    const float* h1w = (const float*)d_in[54]; const float* h1b = (const float*)d_in[55];
    const float* h2w = (const float*)d_in[56]; const float* h2b = (const float*)d_in[57];
    const float* h3w = (const float*)d_in[58]; const float* h3b = (const float*)d_in[59];
    float* out = (float*)d_out;

    // ---- workspace carve-up (fp32) ----
    float* ws   = (float*)d_ws;
    float* s    = ws;                      // [B, 2T, D]       2,097,152
    float* qb   = s   + 2097152;           // [B, Tt, D]       2,097,152
    float* kb   = qb  + 2097152;
    float* vb   = kb  + 2097152;
    float* ob   = vb  + 2097152;
    float* ffn  = ob  + 2097152;           // [B, Tt, DFF]     8,388,608
    float* sc   = ffn + 8388608;           // [B, NH, Tt, Tt] 16,777,216
    float* xIp  = sc  + 16777216;          // [B*T, W]           131,072
    float* xQp  = xIp + 131072;
    float* mbuf = xQp + 131072;            // [B, D]               4,096

    // ---- embedding: de-interleave, conv-as-GEMM, +pos ----
    pack_iq_k<<<(B * L + 255) / 256, 256, 0, stream>>>(xin, xIp, xQp, B * L);
    // tI -> first T tokens of each batch slab of s; tQ -> second T tokens
    G(stream, xIp, W, (long long)T * W, 0,  cIw, D, 0, 0, 0,
      s,            D, SBATCH, 0,  cIb, 1.f, 0,  T, D, W, 1, B);
    G(stream, xQp, W, (long long)T * W, 0,  cQw, D, 0, 0, 0,
      s + T * D,    D, SBATCH, 0,  cQb, 1.f, 0,  T, D, W, 1, B);
    add_pos_k<<<(B * T * D + 255) / 256, 256, 0, stream>>>(s,         SBATCH, pos, T * D, B * T * D);
    add_pos_k<<<(B * T * D + 255) / 256, 256, 0, stream>>>(s + T * D, SBATCH, pos, T * D, B * T * D);

    // ---- encoder runner ----
    auto run_encoder = [&](float* xp, int Tt, const float* const* P, int nl) {
        const long long dd  = (long long)D * D;
        const long long dv  = D;
        const long long wff = (long long)D * DFF;
        const long long td  = (long long)Tt * D;
        const long long tt  = (long long)Tt * Tt;
        const long long tf  = (long long)Tt * DFF;
        for (int l = 0; l < nl; ++l) {
            const float *wq = P[0] + l * dd,  *bq  = P[1]  + l * dv;
            const float *wk = P[2] + l * dd,  *bk_ = P[3]  + l * dv;
            const float *wv = P[4] + l * dd,  *bv_ = P[5]  + l * dv;
            const float *wo = P[6] + l * dd,  *bo  = P[7]  + l * dv;
            const float *w1 = P[8] + l * wff, *bb1 = P[9]  + (long long)l * DFF;
            const float *w2 = P[10]+ l * wff, *bb2 = P[11] + l * dv;
            const float *g1 = P[12]+ l * dv,  *b1  = P[13] + l * dv;
            const float *g2 = P[14]+ l * dv,  *b2  = P[15] + l * dv;

            // QKV projections (batched over B)
            G(stream, xp, D, SBATCH, 0,  wq, D, 0, 0, 0,  qb, D, td, 0, bq,  1.f, 0, Tt, D, D, 1, B);
            G(stream, xp, D, SBATCH, 0,  wk, D, 0, 0, 0,  kb, D, td, 0, bk_, 1.f, 0, Tt, D, D, 1, B);
            G(stream, xp, D, SBATCH, 0,  wv, D, 0, 0, 0,  vb, D, td, 0, bv_, 1.f, 0, Tt, D, D, 1, B);
            // scores = Q Kt / sqrt(dk), batched over (B, NH) via strided head views
            G(stream, qb, D, td, DK,  kb, D, td, DK, 1,
              sc, Tt, (long long)NH * tt, tt, nullptr, 0.125f, 0, Tt, Tt, DK, NH, B * NH);
            softmax_rows_k<<<(B * NH * Tt) / 8, 256, 0, stream>>>(sc, Tt, B * NH * Tt);
            // O_head = P V  (written straight into [B,Tt,D] head slices)
            G(stream, sc, Tt, (long long)NH * tt, tt,  vb, D, td, DK, 0,
              ob, D, td, DK, nullptr, 1.f, 0, Tt, DK, Tt, NH, B * NH);
            // output projection -> qb (tmp), then x = LN(x + o)
            G(stream, ob, D, td, 0,  wo, D, 0, 0, 0,  qb, D, td, 0, bo, 1.f, 0, Tt, D, D, 1, B);
            add_ln_k<<<B * Tt, 128, 0, stream>>>(xp, SBATCH, Tt, qb, g1, b1);
            // FFN: gelu(x w1 + b1) w2 + b2, then x = LN(x + f)
            G(stream, xp, D, SBATCH, 0,  w1, DFF, 0, 0, 0,  ffn, DFF, tf, 0, bb1, 1.f, 1, Tt, DFF, D, 1, B);
            G(stream, ffn, DFF, tf, 0,  w2, D, 0, 0, 0,  ob, D, td, 0, bb2, 1.f, 0, Tt, D, DFF, 1, B);
            add_ln_k<<<B * Tt, 128, 0, stream>>>(xp, SBATCH, Tt, ob, g2, b2);
        }
    };

    run_encoder(s,          T,     encI, 3);   // I branch (first T tokens of each slab)
    run_encoder(s + T * D,  T,     encQ, 3);   // Q branch (second T tokens)
    run_encoder(s,          2 * T, encS, 8);   // fused sequence, 2T tokens

    // ---- mean pool + head ----
    mean_k<<<(B * D + 255) / 256, 256, 0, stream>>>(s, mbuf, 2 * T, SBATCH, B * D);
    head_mlp_k<<<1, 256, 0, stream>>>(mbuf, h1w, h1b, h2w, h2b, h3w, h3b, out);
}
